// QLSTM_57071525429900
// MI455X (gfx1250) — matrix operand order repaired and checked by reading, not compiled, not run on hardware
//
#include <hip/hip_runtime.h>

// Problem dims (fixed by the reference)
#define B_ 16
#define S_ 1024
#define I_ 512
#define H_ 512
#define G_ 2048            // 4*H
#define M_ (B_ * S_)       // 16384 rows of x / hs
#define NQ_ 8
#define DIM_ 256

typedef __attribute__((ext_vector_type(16))) __bf16 v16bf;
typedef __attribute__((ext_vector_type(8)))  __bf16 v8bf;
typedef __attribute__((ext_vector_type(8)))  float  v8f;
typedef __attribute__((ext_vector_type(4)))  unsigned int v4u;
typedef __attribute__((ext_vector_type(8)))  int v8i;
typedef __attribute__((ext_vector_type(4)))  int v4i;

#if defined(__gfx1250__) && __has_builtin(__builtin_amdgcn_tensor_load_to_lds) && \
    __has_builtin(__builtin_amdgcn_s_wait_tensorcnt)
#define HAS_TDM 1
#else
#define HAS_TDM 0
#endif

// ---------------------------------------------------------------- helpers ---

static __device__ __forceinline__ v8f wmma_bf16(v16bf a, v16bf b, v8f c) {
  // D = A(16x32 bf16) * B(32x16 bf16) + C(f32)
  return __builtin_amdgcn_wmma_f32_16x16x32_bf16(
      /*neg_a=*/false, a, /*neg_b=*/false, b,
      /*c_mod=*/(short)0, c, /*reuse_a=*/false, /*reuse_b=*/false);
}

// A fragment: row pointer `rowp` = &A[m][0] (m = lane&15), K-chunk base kb.
// ISA layout: lanes 0-15 hold K = kb+[0..7] and kb+16+[0..7]; lanes 16-31 are +8.
static __device__ __forceinline__ v16bf load_a_frag(const __bf16* rowp, int kb, int lane) {
  int ako = (lane < 16) ? 0 : 8;
  v8bf lo = *(const v8bf*)(rowp + kb + ako);
  v8bf hi = *(const v8bf*)(rowp + kb + 16 + ako);
  v16bf f;
#pragma unroll
  for (int i = 0; i < 8; ++i) { f[i] = lo[i]; f[i + 8] = hi[i]; }
  return f;
}

// B fragment: `rowp` = &W[n][0] with n = tile*16 + (lane&15); B[k][n] = W[n][k].
// Lanes 0-15 hold K = kb+[0..7] and kb+16+[0..7]; lanes 16-31 are +8 (same as A).
static __device__ __forceinline__ v16bf load_b_frag(const __bf16* rowp, int kb, int lane) {
  int bko = (lane < 16) ? 0 : 8;
  v8bf lo = *(const v8bf*)(rowp + kb + bko);
  v8bf hi = *(const v8bf*)(rowp + kb + 16 + bko);
  v16bf f;
#pragma unroll
  for (int i = 0; i < 8; ++i) { f[i] = lo[i]; f[i + 8] = hi[i]; }
  return f;
}

static __device__ __forceinline__ float sigmoid_(float x) {
  return 1.0f / (1.0f + __expf(-x));
}
static __device__ __forceinline__ float tanh_(float x) {
  float e = __expf(2.0f * x);
  return 1.0f - 2.0f / (e + 1.0f);
}

#if HAS_TDM
// Tensor Data Mover: 2-D bf16 tile (rows x cols), row-major with `stride`
// elements between rows, global -> LDS at byte offset lds_off.
// D# per CDNA5 ISA.
static __device__ __forceinline__ void tdm_load_tile_bf16(
    unsigned lds_off, const __bf16* gsrc, unsigned rows, unsigned cols,
    unsigned stride) {
  unsigned long long ga = (unsigned long long)(size_t)(const void*)gsrc;
  v4u g0;
  g0[0] = 1u;                                   // count=1, user descriptor
  g0[1] = lds_off;                              // lds_addr
  g0[2] = (unsigned)(ga & 0xffffffffull);       // global_addr[31:0]
  g0[3] = (unsigned)((ga >> 32) & 0x01ffffffull) | (2u << 30);  // addr hi | type=2
  v8i g1;
  g1[0] = (int)(1u << 16);                      // wg_mask=0 | data_size=1 (2 bytes)
  g1[1] = (int)((cols & 0xffffu) << 16);        // tensor_dim0[15:0]
  g1[2] = (int)(((cols >> 16) & 0xffffu) | ((rows & 0xffffu) << 16));
  g1[3] = (int)(((rows >> 16) & 0xffffu) | ((cols & 0xffffu) << 16)); // tile_dim0
  g1[4] = (int)(rows & 0xffffu);                // tile_dim1 | tile_dim2=0
  g1[5] = (int)stride;                          // tensor_dim0_stride[31:0]
  g1[6] = 0;                                    // stride hi | dim1 stride lo
  g1[7] = 0;
  v4i gz = {0, 0, 0, 0};
#if __clang_major__ >= 23
  v8i gz8 = {0, 0, 0, 0, 0, 0, 0, 0};
  __builtin_amdgcn_tensor_load_to_lds(g0, g1, gz, gz, gz8, 0);
#else
  __builtin_amdgcn_tensor_load_to_lds(g0, g1, gz, gz, 0);
#endif
}
#endif

// Grid-wide arrive/spin barrier (persistent LSTM kernel).
static __device__ __forceinline__ void grid_barrier(unsigned* cnt, unsigned* gen,
                                                    unsigned nblocks) {
  __syncthreads();
  if (threadIdx.x == 0) {
    unsigned g = *(volatile unsigned*)gen;
    __threadfence();
    unsigned arrived = atomicAdd(cnt, 1u);
    if (arrived == nblocks - 1u) {
      *(volatile unsigned*)cnt = 0u;
      __threadfence();
      atomicAdd(gen, 1u);
    } else {
      while (*(volatile unsigned*)gen == g) { __builtin_amdgcn_s_sleep(1); }
      __threadfence();
    }
  }
  __syncthreads();
}

// ---------------------------------------------------------------- kernels ---

__global__ void init_kernel(unsigned* bar, __bf16* hbuf, int nh) {
  int i = blockIdx.x * blockDim.x + threadIdx.x;
  if (i < 2) bar[i] = 0u;
  if (i < nh) hbuf[i] = (__bf16)0.0f;
}

__global__ void cvt_bf16_kernel(const float* __restrict__ src,
                                __bf16* __restrict__ dst, int n) {
  int i = blockIdx.x * blockDim.x + threadIdx.x;
  if (i < n) dst[i] = (__bf16)src[i];
}

__global__ void bias_kernel(const float* __restrict__ b_ih,
                            const float* __restrict__ b_hh,
                            float* __restrict__ bias, int n) {
  int i = blockIdx.x * blockDim.x + threadIdx.x;
  if (i < n) bias[i] = b_ih[i] + b_hh[i];
}

// xp[M_, G_] = x_bf16[M_, I_] @ W_ih_bf16[G_, I_]^T + bias[G_]
// Block = 8 waves sharing one 16-row x tile (TDM-staged into LDS).
// Wave -> 16(M) x 64(N); K pipelined in 32-chunks with 2x unrolled ping-pong
// fragment sets (no register-rotation movs, no WMMA->VALU hazards).
__global__ __launch_bounds__(256) void gemm_xp_kernel(
    const __bf16* __restrict__ Xb, const __bf16* __restrict__ Wb,
    const float* __restrict__ bias, float* __restrict__ xp) {
  __shared__ __bf16 sx[16 * I_];                 // 16 KB x tile
  int lane = threadIdx.x & 31;
  int wv = threadIdx.x >> 5;
  int mt = blockIdx.x >> 2;                      // 1024 M-tiles, 4 blocks each
  int ng = (blockIdx.x & 3) * 8 + wv;            // 32 N-groups of 64

  // stage x tile (16 x 512 bf16) into LDS
#if HAS_TDM
  if (wv == 0) {
    tdm_load_tile_bf16((unsigned)(size_t)(void*)sx, Xb + mt * 16 * I_, 16, I_, I_);
    __builtin_amdgcn_s_wait_tensorcnt(0);
  }
#else
  for (int i = threadIdx.x * 8; i < 16 * I_; i += 256 * 8)
    *(v8bf*)(sx + i) = *(const v8bf*)(Xb + mt * 16 * I_ + i);
#endif
  __syncthreads();

  int bcol = lane & 15;
  const __bf16* arowp = sx + (lane & 15) * I_;
  const __bf16* brow[4];
#pragma unroll
  for (int t = 0; t < 4; ++t) brow[t] = Wb + (ng * 64 + t * 16 + bcol) * I_;

  v8f acc[4] = {v8f{}, v8f{}, v8f{}, v8f{}};
  v16bf aA = load_a_frag(arowp, 0, lane);
  v16bf bA[4];
#pragma unroll
  for (int t = 0; t < 4; ++t) bA[t] = load_b_frag(brow[t], 0, lane);

#pragma unroll 1
  for (int kc = 0; kc < I_ / 32; kc += 2) {
    int k1 = (kc + 1) * 32;                      // kc+1 <= 15 always
    v16bf aB = load_a_frag(arowp, k1, lane);
    v16bf bB[4];
#pragma unroll
    for (int t = 0; t < 4; ++t) bB[t] = load_b_frag(brow[t], k1, lane);
#pragma unroll
    for (int t = 0; t < 4; ++t) acc[t] = wmma_bf16(aA, bA[t], acc[t]);
    int k2 = (kc + 2 < I_ / 32) ? (kc + 2) * 32 : 0;  // dummy reload on last iter
    aA = load_a_frag(arowp, k2, lane);
#pragma unroll
    for (int t = 0; t < 4; ++t) bA[t] = load_b_frag(brow[t], k2, lane);
#pragma unroll
    for (int t = 0; t < 4; ++t) acc[t] = wmma_bf16(aB, bB[t], acc[t]);
  }

  int mbase = mt * 16 + ((lane < 16) ? 0 : 8);
#pragma unroll
  for (int t = 0; t < 4; ++t) {
    int col = ng * 64 + t * 16 + bcol;
    float bv = bias[col];
#pragma unroll
    for (int r = 0; r < 8; ++r) {
      xp[(mbase + r) * G_ + col] = acc[t][r] + bv;
    }
  }
}

// Persistent LSTM: 8 blocks x 128 threads = 32 waves; wave w owns hidden tile w
// (16 hidden units) and its 4 gate tiles; cell state c lives in registers.
// Per step: h_prev (16x512 bf16) TDM-staged into LDS, gates via WMMA, grid sync.
#define LSTM_BLOCKS 8
__global__ __launch_bounds__(128) void lstm_kernel(
    const __bf16* __restrict__ Whh, const float* __restrict__ xp,
    __bf16* __restrict__ hbuf, float* __restrict__ hs, unsigned* bar) {
  __shared__ __bf16 sh[B_ * H_];                 // 16 KB h tile
  int lane = threadIdx.x & 31;
  int wv = threadIdx.x >> 5;
  int ht = blockIdx.x * 4 + wv;                  // hidden tile 0..31
  int bcol = lane & 15;
  int mb = (lane < 16) ? 0 : 8;                  // C/D row offset
  int hcol = ht * 16 + bcol;
  const __bf16* arowp = sh + (lane & 15) * H_;
  const __bf16* brow[4];
#pragma unroll
  for (int g = 0; g < 4; ++g) brow[g] = Whh + (g * H_ + ht * 16 + bcol) * H_;

  v8f c = v8f{};
  for (int t = 0; t < S_; ++t) {
    // stage h_{t-1} into LDS (grid barrier at previous step end makes it safe)
#if HAS_TDM
    if (wv == 0) {
      tdm_load_tile_bf16((unsigned)(size_t)(void*)sh,
                         hbuf + (t & 1) * (B_ * H_), B_, H_, H_);
      __builtin_amdgcn_s_wait_tensorcnt(0);
    }
#else
    {
      const __bf16* src = hbuf + (t & 1) * (B_ * H_);
      for (int i = threadIdx.x * 8; i < B_ * H_; i += 128 * 8)
        *(v8bf*)(sh + i) = *(const v8bf*)(src + i);
    }
#endif
    __syncthreads();

    // prefetch next step's xp gate values (HBM stream) while we compute
    if (t + 1 < S_) {
      int basep = (mb * S_ + (t + 1)) * G_;
#pragma unroll
      for (int g = 0; g < 4; ++g)
        __builtin_prefetch(&xp[basep + g * H_ + hcol], 0, 1);
    }

    v8f acc[4] = {v8f{}, v8f{}, v8f{}, v8f{}};
    v16bf aA = load_a_frag(arowp, 0, lane);
    v16bf bA[4];
#pragma unroll
    for (int g = 0; g < 4; ++g) bA[g] = load_b_frag(brow[g], 0, lane);

#pragma unroll 1
    for (int kc = 0; kc < H_ / 32; kc += 2) {
      int k1 = (kc + 1) * 32;
      v16bf aB = load_a_frag(arowp, k1, lane);
      v16bf bB[4];
#pragma unroll
      for (int g = 0; g < 4; ++g) bB[g] = load_b_frag(brow[g], k1, lane);
#pragma unroll
      for (int g = 0; g < 4; ++g) acc[g] = wmma_bf16(aA, bA[g], acc[g]);
      int k2 = (kc + 2 < H_ / 32) ? (kc + 2) * 32 : 0;
      aA = load_a_frag(arowp, k2, lane);
#pragma unroll
      for (int g = 0; g < 4; ++g) bA[g] = load_b_frag(brow[g], k2, lane);
#pragma unroll
      for (int g = 0; g < 4; ++g) acc[g] = wmma_bf16(aB, bB[g], acc[g]);
    }

    // gates + cell update (i,f,g,o blocks of 512 in the 4H dim)
    v8f hv;
#pragma unroll
    for (int r = 0; r < 8; ++r) {
      int bb = mb + r;
      int base = (bb * S_ + t) * G_;
      float gi = sigmoid_(acc[0][r] + xp[base + 0 * H_ + hcol]);
      float gf = sigmoid_(acc[1][r] + xp[base + 1 * H_ + hcol]);
      float gg = tanh_   (acc[2][r] + xp[base + 2 * H_ + hcol]);
      float go = sigmoid_(acc[3][r] + xp[base + 3 * H_ + hcol]);
      float cn = gf * c[r] + gi * gg;
      c[r] = cn;
      hv[r] = go * tanh_(cn);
    }
    __bf16* hnext = hbuf + ((t + 1) & 1) * (B_ * H_);
#pragma unroll
    for (int r = 0; r < 8; ++r) {
      int bb = mb + r;
      hnext[bb * H_ + hcol] = (__bf16)hv[r];
      hs[(bb * S_ + t) * H_ + hcol] = hv[r];
    }
    grid_barrier(bar, bar + 1, LSTM_BLOCKS);
  }
}

// Fused tail: one wave per sample n.
//   z = hs[n]@Wq^T + bq  ->  8-qubit circuit (RX embed, 2x[ISWAP chain + RZ diag])
//   -> <Z_q>  ->  out[n] = q@Wh^T + bh
#define TAIL_WAVES 8
__global__ __launch_bounds__(256) void tail_kernel(
    const float* __restrict__ hs, const float* __restrict__ Wq,
    const float* __restrict__ bq, const float* __restrict__ qw,
    const float* __restrict__ Wh, const float* __restrict__ bh,
    float* __restrict__ out) {
  __shared__ float sWq[NQ_ * H_];
  __shared__ float sWh[H_ * NQ_];
  __shared__ float sbh[H_];
  __shared__ float sbq[NQ_];
  __shared__ float sqw[NQ_];
  __shared__ float amp[TAIL_WAVES][2 * DIM_];   // [wave][re 0..255 | im 256..511]

  for (int i = threadIdx.x; i < NQ_ * H_; i += blockDim.x) sWq[i] = Wq[i];
  for (int i = threadIdx.x; i < H_ * NQ_; i += blockDim.x) sWh[i] = Wh[i];
  for (int i = threadIdx.x; i < H_; i += blockDim.x) sbh[i] = bh[i];
  if (threadIdx.x < NQ_) { sbq[threadIdx.x] = bq[threadIdx.x]; sqw[threadIdx.x] = qw[threadIdx.x]; }
  __syncthreads();

  int lane = threadIdx.x & 31;
  int wv = threadIdx.x >> 5;
  int n = blockIdx.x * TAIL_WAVES + wv;          // exact: 2048 blocks * 8 = 16384

  // ---- z = hs[n] @ Wq^T + bq  (wave-cooperative dot products) ----
  const float* hrow = hs + n * H_ + lane * 16;
  float hvv[16];
#pragma unroll
  for (int j = 0; j < 16; ++j) hvv[j] = hrow[j];
  float z[NQ_];
#pragma unroll
  for (int q = 0; q < NQ_; ++q) {
    float p = 0.f;
    const float* wq = sWq + q * H_ + lane * 16;
#pragma unroll
    for (int j = 0; j < 16; ++j) p += hvv[j] * wq[j];
#pragma unroll
    for (int off = 16; off > 0; off >>= 1) p += __shfl_xor(p, off, 32);
    z[q] = p + sbq[q];
  }

  // ---- RX product state: amp[i] = prod_q (bit? -i*sin : cos), bit=(i>>(7-q))&1
  float cq[NQ_], sq[NQ_];
#pragma unroll
  for (int q = 0; q < NQ_; ++q) __sincosf(0.5f * z[q], &sq[q], &cq[q]);

  float re[8], im[8], dre[8], dim_[8];
#pragma unroll
  for (int s = 0; s < 8; ++s) {
    int idx = s * 32 + lane;
    float r = 1.f, m = 0.f, ang = 0.f;
#pragma unroll
    for (int q = 0; q < NQ_; ++q) {
      int bit = (idx >> (7 - q)) & 1;
      float fr = bit ? 0.f : cq[q];
      float fi = bit ? -sq[q] : 0.f;
      float nr = r * fr - m * fi;
      float ni = r * fi + m * fr;
      r = nr; m = ni;
      ang += bit ? 0.5f * sqw[q] : -0.5f * sqw[q];   // exp(-0.5i*sign*w)
    }
    re[s] = r; im[s] = m;
    __sincosf(ang, &dim_[s], &dre[s]);
  }

  float* are = &amp[wv][0];
  float* aim = &amp[wv][DIM_];
  for (int layer = 0; layer < 2; ++layer) {
#pragma unroll
    for (int q = 0; q < NQ_ - 1; ++q) {            // ISWAP on wires q,q+1
      int hi = 7 - q, lo = 6 - q;
      int mask = (1 << hi) | (1 << lo);
#pragma unroll
      for (int s = 0; s < 8; ++s) { int i = s * 32 + lane; are[i] = re[s]; aim[i] = im[s]; }
      asm volatile("" ::: "memory");               // keep DS store->load order
#pragma unroll
      for (int s = 0; s < 8; ++s) {
        int i = s * 32 + lane;
        int b1 = (i >> hi) & 1, b0 = (i >> lo) & 1;
        if (b1 != b0) {                            // |01><10| * i
          int j = i ^ mask;
          float orr = are[j], oii = aim[j];
          re[s] = -oii; im[s] = orr;
        } else {
          re[s] = are[i]; im[s] = aim[i];
        }
      }
      asm volatile("" ::: "memory");
    }
#pragma unroll
    for (int s = 0; s < 8; ++s) {                  // RZ diagonal
      float r = re[s], m = im[s];
      re[s] = r * dre[s] - m * dim_[s];
      im[s] = r * dim_[s] + m * dre[s];
    }
  }

  // ---- probs -> <Z_q> ----
  float ex[NQ_] = {0.f, 0.f, 0.f, 0.f, 0.f, 0.f, 0.f, 0.f};
#pragma unroll
  for (int s = 0; s < 8; ++s) {
    int idx = s * 32 + lane;
    float p = re[s] * re[s] + im[s] * im[s];
#pragma unroll
    for (int q = 0; q < NQ_; ++q) ex[q] += ((idx >> (7 - q)) & 1) ? -p : p;
  }
#pragma unroll
  for (int q = 0; q < NQ_; ++q)
#pragma unroll
    for (int off = 16; off > 0; off >>= 1) ex[q] += __shfl_xor(ex[q], off, 32);

  // ---- out[n] = ex @ Wh^T + bh ----
  float* orow = out + n * H_;
#pragma unroll
  for (int jj = 0; jj < 16; ++jj) {
    int j = lane * 16 + jj;
    float a = sbh[j];
#pragma unroll
    for (int q = 0; q < NQ_; ++q) a += ex[q] * sWh[j * NQ_ + q];
    orow[j] = a;
  }
}

// ---------------------------------------------------------------- launch ----

extern "C" void kernel_launch(void* const* d_in, const int* in_sizes, int n_in,
                              void* d_out, int out_size, void* d_ws, size_t ws_size,
                              hipStream_t stream) {
  (void)in_sizes; (void)n_in; (void)out_size; (void)ws_size;
  const float* x    = (const float*)d_in[0];
  const float* W_ih = (const float*)d_in[1];
  const float* W_hh = (const float*)d_in[2];
  const float* b_ih = (const float*)d_in[3];
  const float* b_hh = (const float*)d_in[4];
  const float* Wq   = (const float*)d_in[5];
  const float* bq   = (const float*)d_in[6];
  const float* qw   = (const float*)d_in[7];
  const float* Wh   = (const float*)d_in[8];
  const float* bh   = (const float*)d_in[9];
  float* out = (float*)d_out;

  char* ws = (char*)d_ws;
  size_t off = 0;
  float*  xp      = (float*) (ws + off); off += (size_t)M_ * G_ * 4;   // 128 MB
  float*  hs      = (float*) (ws + off); off += (size_t)M_ * H_ * 4;   // 32 MB
  __bf16* x_bf    = (__bf16*)(ws + off); off += (size_t)M_ * I_ * 2;   // 16 MB
  __bf16* Wih_bf  = (__bf16*)(ws + off); off += (size_t)G_ * I_ * 2;   // 2 MB
  __bf16* Whh_bf  = (__bf16*)(ws + off); off += (size_t)G_ * H_ * 2;   // 2 MB
  float*  bias    = (float*) (ws + off); off += (size_t)G_ * 4;
  __bf16* hbuf    = (__bf16*)(ws + off); off += (size_t)2 * B_ * H_ * 2;
  unsigned* bar   = (unsigned*)(ws + off); off += 256;

  // init barrier + zero h
  init_kernel<<<(2 * B_ * H_ + 255) / 256, 256, 0, stream>>>(bar, hbuf, 2 * B_ * H_);

  // bf16 conversions + fused bias
  cvt_bf16_kernel<<<(M_ * I_) / 256, 256, 0, stream>>>(x, x_bf, M_ * I_);
  cvt_bf16_kernel<<<(G_ * I_) / 256, 256, 0, stream>>>(W_ih, Wih_bf, G_ * I_);
  cvt_bf16_kernel<<<(G_ * H_) / 256, 256, 0, stream>>>(W_hh, Whh_bf, G_ * H_);
  bias_kernel<<<G_ / 256, 256, 0, stream>>>(b_ih, b_hh, bias, G_);

  // xp = x @ W_ih^T + (b_ih + b_hh)   [WMMA bf16, x tile via TDM->LDS]
  gemm_xp_kernel<<<(M_ / 16) * 4, 256, 0, stream>>>(x_bf, Wih_bf, bias, xp);

  // persistent LSTM over S=1024 steps [WMMA bf16 + TDM h-stage + grid barrier]
  lstm_kernel<<<LSTM_BLOCKS, 128, 0, stream>>>(Whh_bf, xp, hbuf, hs, bar);

  // fused z -> quantum -> output
  tail_kernel<<<M_ / TAIL_WAVES, 256, 0, stream>>>(hs, Wq, bq, qw, Wh, bh, out);
}
